// SpDHRGAT_84954453115144
// MI455X (gfx1250) — matrix operand order repaired and checked by reading, not compiled, not run on hardware
//
#include <hip/hip_runtime.h>
#include <hip/hip_bf16.h>
#include <stdint.h>

// ---------------------------------------------------------------------------
// SpDHRGAT for MI455X (gfx1250, wave32).
//
//  * theta depends only on edge_type (R=200) -> per-relation MLP, not per-edge.
//  * logits = s[row,h] + (x[row]·(cosθ_etype ⊙ u_h) + c2_h) + t[col,h]
//  * W_h_j per edge = gather of y = x @ Wcat  (WMMA GEMM [N,128]@[128,256]).
//  * fuse GEMM [N,512]@[512,256] + elu via WMMA with split A = (h_local | Hg).
//
// GEMM: block = 8 waves computes 128x64 tile; per 32-K step a 128x32 f32
// A-panel is staged in LDS (coalesced float2 loads, pad-33 banking), each
// wave builds one bf16 A fragment and issues 4 v_wmma_f32_16x16x32_bf16
// (f32 accumulate). K is a template param so the K loop fully unrolls.
// ---------------------------------------------------------------------------

#define FDIM 128
#define HD   256   // H*D
#define HEADS 4

typedef __bf16 bf16;
typedef __attribute__((ext_vector_type(16))) __bf16 v16bf;
typedef __attribute__((ext_vector_type(8)))  float  v8f;

union Frag { v16bf v; uint32_t u[8]; };

static __device__ __forceinline__ uint32_t pack_bf16(float lo, float hi) {
  bf16 l = (bf16)lo, h = (bf16)hi;
  return ((uint32_t)__builtin_bit_cast(unsigned short, h) << 16) |
         (uint32_t)__builtin_bit_cast(unsigned short, l);
}

// ---------------- small precompute kernels ----------------

// v1[h][f]=Σ_d Wh·a1, u[h][f]=Σ_d Wr·a2, v3[h][f]=Σ_d Wh·a3 ; c2[h]=Σ_f cr·u
__global__ void k_prevec(const float* __restrict__ Wh, const float* __restrict__ Wr,
                         const float* __restrict__ a, const float* __restrict__ cr,
                         float* __restrict__ VEC, float* __restrict__ c2) {
  int h = blockIdx.x, f = threadIdx.x;  // 128 threads
  const float* Whh = Wh + (size_t)h * FDIM * 64;
  const float* Wrh = Wr + (size_t)h * FDIM * 64;
  const float* ah  = a + (size_t)h * 192;
  float s1 = 0.f, s2 = 0.f, s3 = 0.f;
  for (int d = 0; d < 64; ++d) {
    float w = Whh[f * 64 + d], wr = Wrh[f * 64 + d];
    s1 += w * ah[d]; s2 += wr * ah[64 + d]; s3 += w * ah[128 + d];
  }
  VEC[(h * 3 + 0) * FDIM + f] = s1;
  VEC[(h * 3 + 1) * FDIM + f] = s2;
  VEC[(h * 3 + 2) * FDIM + f] = s3;
  __shared__ float red[FDIM];
  red[f] = cr[h * FDIM + f] * s2;
  __syncthreads();
  for (int s = 64; s > 0; s >>= 1) { if (f < s) red[f] += red[f + s]; __syncthreads(); }
  if (f == 0) c2[h] = red[0];
}

// Per (relation, head): theta = relu(rel@W1+b1)@W2+b2 ; wvec = cos(theta)*u
__global__ void k_relmlp(const float* __restrict__ rel, const float* __restrict__ W1,
                         const float* __restrict__ b1, const float* __restrict__ W2,
                         const float* __restrict__ b2, const float* __restrict__ VEC,
                         float* __restrict__ wvec, int R) {
  int r = blockIdx.x, h = blockIdx.y, tid = threadIdx.x;  // 256 threads
  __shared__ float srel[FDIM], shid[512];
  if (tid < FDIM) srel[tid] = rel[(size_t)r * FDIM + tid];
  __syncthreads();
  const float* W1h = W1 + (size_t)h * FDIM * 512;
  for (int j = tid; j < 512; j += 256) {
    float acc = b1[h * 512 + j];
    for (int f = 0; f < FDIM; ++f) acc += srel[f] * W1h[(size_t)f * 512 + j];
    shid[j] = fmaxf(acc, 0.f);
  }
  __syncthreads();
  const float* W2h = W2 + (size_t)h * 512 * FDIM;
  const float* u = VEC + (h * 3 + 1) * FDIM;
  for (int f = tid; f < FDIM; f += 256) {
    float acc = b2[h * FDIM + f];
    for (int j = 0; j < 512; ++j) acc += shid[j] * W2h[(size_t)j * FDIM + f];
    wvec[((size_t)(h * R + r)) * FDIM + f] = cosf(acc) * u[f];
  }
}

// s[i,h]=x[i]·v1_h, t[i,h]=x[i]·v3_h  -> st[i][8] = {s0,t0,s1,t1,...} (wave/node)
__global__ void k_node_scalars(const float* __restrict__ x, const float* __restrict__ VEC,
                               float* __restrict__ st, int N) {
  int lane = threadIdx.x & 31;
  int gw = (blockIdx.x * blockDim.x + threadIdx.x) >> 5;
  int nw = (gridDim.x * blockDim.x) >> 5;
  for (int i = gw; i < N; i += nw) {
    float xr[4];
#pragma unroll
    for (int j = 0; j < 4; ++j) xr[j] = x[(size_t)i * FDIM + lane + 32 * j];
#pragma unroll
    for (int k = 0; k < 8; ++k) {
      int h = k >> 1, which = (k & 1) ? 2 : 0;
      const float* v = VEC + (h * 3 + which) * FDIM;
      float p = 0.f;
#pragma unroll
      for (int j = 0; j < 4; ++j) p += xr[j] * v[lane + 32 * j];
#pragma unroll
      for (int off = 16; off > 0; off >>= 1) p += __shfl_xor(p, off);
      if (lane == k) st[(size_t)i * 8 + k] = p;
    }
  }
}

// ---------------- weight packing (f32 -> bf16, K-pair interleaved) ----------
// Bp[(k>>1)*Nc + n] holds bf16 pair {B[k_even][n], B[k_odd][n]}.

__global__ void k_pack_rowmajor(const float* __restrict__ B, uint32_t* __restrict__ Bp,
                                int K, int Nc) {
  int idx = blockIdx.x * blockDim.x + threadIdx.x;
  int tot = (K >> 1) * Nc;
  if (idx >= tot) return;
  int n = idx % Nc, k2 = idx / Nc;
  Bp[idx] = pack_bf16(B[(size_t)(2 * k2) * Nc + n], B[(size_t)(2 * k2 + 1) * Nc + n]);
}

// Wcat[f][h*64+d] = head_W[h][f][d]
__global__ void k_pack_headW(const float* __restrict__ W, uint32_t* __restrict__ Bp) {
  int idx = blockIdx.x * blockDim.x + threadIdx.x;  // (128/2)*256
  if (idx >= 64 * HD) return;
  int n = idx % HD, f2 = idx / HD;
  int h = n >> 6, d = n & 63;
  float lo = W[((size_t)h * FDIM + 2 * f2) * 64 + d];
  float hi = W[((size_t)h * FDIM + 2 * f2 + 1) * 64 + d];
  Bp[idx] = pack_bf16(lo, hi);
}

// ---------------- WMMA GEMM: C[M,Nc] = [A0|A1] @ B (+bias, optional elu) ----
// Block = 256 thr = 8 waves -> 128x64 output tile; wave w owns rows 16w..16w+15
// and all 4 column sub-tiles. Per K-step: stage 128x32 f32 A-panel in LDS
// (coalesced), build one A fragment per wave, 4 WMMAs. K0/K1 compile-time.
template <int K0, int K1>
__global__ void k_wmma_gemm_t(const float* __restrict__ A0,
                              const float* __restrict__ A1,
                              const uint32_t* __restrict__ Bp,
                              const float* __restrict__ bias,
                              float* __restrict__ C, int M, int Nc, int act) {
  constexpr int KT = K0 + K1;
  __shared__ float As[128][33];          // pad 33: conflict-free frag reads
  int tid = threadIdx.x;
  int lane = tid & 31, wv = tid >> 5;
  int half = lane >> 4, r = lane & 15;
  int mtiles = (M + 127) >> 7;
  int ntiles = Nc >> 6;
  int tot = mtiles * ntiles;
  for (int bt = blockIdx.x; bt < tot; bt += gridDim.x) {
    int bm = (bt / ntiles) * 128;
    int bn0 = (bt % ntiles) * 64;
    v8f acc[4] = {{}, {}, {}, {}};
#pragma unroll
    for (int kk = 0; kk < KT; kk += 32) {
      const float* Asrc = (kk < K0) ? A0 : A1;      // folds after unroll
      const int lda  = (kk < K0) ? K0 : K1;
      const int kloc = (kk < K0) ? kk : kk - K0;
      __syncthreads();
      // stage 128 rows x 32 K of f32: 2048 float2 over 256 threads
#pragma unroll
      for (int i = 0; i < 8; ++i) {
        int flat = tid + 256 * i;                    // 0..2047
        int row = flat >> 4, kp = (flat & 15) * 2;
        int gr = bm + row; if (gr >= M) gr = M - 1;  // clamp; stores guarded
        float2 v2 = *(const float2*)&Asrc[(size_t)gr * lda + kloc + kp];
        As[row][kp]     = v2.x;
        As[row][kp + 1] = v2.y;
      }
      __syncthreads();
      Frag a;
#pragma unroll
      for (int v = 0; v < 8; ++v) {
        // A 16x32 bf16 layout: lanes 0-15 rows, K = {2v,2v+1}(+16 v>=4)+8*half
        int ka = ((v & 3) * 2) + ((v >> 2) * 16) + half * 8;
        a.u[v] = pack_bf16(As[wv * 16 + r][ka], As[wv * 16 + r][ka + 1]);
      }
#pragma unroll
      for (int j = 0; j < 4; ++j) {
        Frag b;
#pragma unroll
        for (int v = 0; v < 8; ++v) {
          // B 32x16 bf16: K = {2v,2v+1}(+16 v>=4)+8*half, same as A
          int kb2 = (kk >> 1) + (v & 3) + (v >> 2) * 8 + 4 * half;  // packed-pair index
          b.u[v] = Bp[(size_t)kb2 * Nc + bn0 + 16 * j + r];
        }
        acc[j] = __builtin_amdgcn_wmma_f32_16x16x32_bf16(
            false, a.v, false, b.v, (short)0, acc[j], false, false);
      }
    }
#pragma unroll
    for (int j = 0; j < 4; ++j) {
#pragma unroll
      for (int i = 0; i < 8; ++i) {
        int m = bm + wv * 16 + i + 8 * half;
        if (m < M) {
          int n = bn0 + 16 * j + r;
          float vv = acc[j][i] + (bias ? bias[n] : 0.f);
          if (act) vv = vv > 0.f ? vv : expm1f(vv);
          C[(size_t)m * Nc + n] = vv;
        }
      }
    }
  }
}

// ---------------- edge kernels (wave32 per edge) ----------------

__global__ void k_edges(const float* __restrict__ x, const int* __restrict__ row,
                        const int* __restrict__ col, const int* __restrict__ et,
                        const float* __restrict__ wvec, const float* __restrict__ st,
                        const float* __restrict__ c2, const float* __restrict__ y,
                        float* __restrict__ rowsum, float* __restrict__ num,
                        int E, int R) {
  int lane = threadIdx.x & 31;
  int gw = (blockIdx.x * blockDim.x + threadIdx.x) >> 5;
  int nw = (gridDim.x * blockDim.x) >> 5;
  for (int e = gw; e < E; e += nw) {
    int ri = row[e], cj = col[e], t = et[e];
    int e2 = e + nw;
    if (e2 < E) __builtin_prefetch(&x[(size_t)row[e2] * FDIM], 0, 0);
    const float* xr = x + (size_t)ri * FDIM;
    float xl[4];
#pragma unroll
    for (int j = 0; j < 4; ++j) xl[j] = xr[lane + 32 * j];
    float eh[4];
#pragma unroll
    for (int h = 0; h < HEADS; ++h) {
      const float* w = wvec + ((size_t)(h * R + t)) * FDIM;
      float p = 0.f;
#pragma unroll
      for (int j = 0; j < 4; ++j) p += xl[j] * w[lane + 32 * j];
#pragma unroll
      for (int off = 16; off > 0; off >>= 1) p += __shfl_xor(p, off);
      float logit = st[(size_t)ri * 8 + 2 * h] + p + c2[h] + st[(size_t)cj * 8 + 2 * h + 1];
      float lr = logit > 0.f ? logit : 0.2f * logit;     // leaky_relu(0.2)
      eh[h] = __expf(-lr);
    }
    if (lane < 4) {
      float esel = lane == 0 ? eh[0] : lane == 1 ? eh[1] : lane == 2 ? eh[2] : eh[3];
      atomicAdd(&rowsum[(size_t)ri * 4 + lane], esel);
    }
    const float* yc = y + (size_t)cj * HD;
    float* nr = num + (size_t)ri * HD;
#pragma unroll
    for (int it = 0; it < 8; ++it) {
      int cidx = lane + 32 * it;         // head = it>>1 (compile-time per iter)
      atomicAdd(&nr[cidx], eh[it >> 1] * yc[cidx]);
    }
  }
}

__global__ void k_count_deg(const int* __restrict__ row, float* __restrict__ deg, int E) {
  int e = blockIdx.x * blockDim.x + threadIdx.x;
  if (e < E) atomicAdd(&deg[row[e]], 1.0f);
}

__global__ void k_dinv(const float* __restrict__ deg, float* __restrict__ dinv, int N) {
  int i = blockIdx.x * blockDim.x + threadIdx.x;
  if (i < N) dinv[i] = rsqrtf(deg[i] + 1.0f);   // +1 self loop; always > 0
}

__global__ void k_finalize_local(float* __restrict__ num, const float* __restrict__ rowsum,
                                 int N) {
  int idx = blockIdx.x * blockDim.x + threadIdx.x;
  if (idx >= N * HD) return;
  int i = idx >> 8, h = (idx >> 6) & 3;
  float rs = rowsum[(size_t)i * 4 + h];
  if (rs == 0.f) rs = 1e-12f;
  float v = num[idx] / rs;
  num[idx] = v > 0.f ? v : expm1f(v);            // elu, in place -> h_local
}

// dst = dinv[i]^2 * src  (self-loop term; doubles as the hop's initialization)
__global__ void k_hop_init(float* __restrict__ dst, const float* __restrict__ src,
                           const float* __restrict__ dinv, int N) {
  int idx = blockIdx.x * blockDim.x + threadIdx.x;
  if (idx >= N * HD) return;
  int i = idx >> 8;
  float d = dinv[i];
  dst[idx] = d * d * src[idx];
}

__global__ void k_hop_edges(const int* __restrict__ row, const int* __restrict__ col,
                            const float* __restrict__ dinv, float* __restrict__ dst,
                            const float* __restrict__ src, int E) {
  int lane = threadIdx.x & 31;
  int gw = (blockIdx.x * blockDim.x + threadIdx.x) >> 5;
  int nw = (gridDim.x * blockDim.x) >> 5;
  for (int e = gw; e < E; e += nw) {
    int ri = row[e], cj = col[e];
    float coef = dinv[ri] * dinv[cj];
    float* dr = dst + (size_t)ri * HD;
    const float* sc = src + (size_t)cj * HD;
#pragma unroll
    for (int it = 0; it < 8; ++it) {
      int cidx = lane + 32 * it;
      atomicAdd(&dr[cidx], coef * sc[cidx]);
    }
  }
}

__global__ void k_zero(float* __restrict__ p, long n) {
  long i = (long)blockIdx.x * blockDim.x + threadIdx.x;
  long stride = (long)gridDim.x * blockDim.x;
  for (; i < n; i += stride) p[i] = 0.f;
}

// ---------------- host orchestration ----------------

extern "C" void kernel_launch(void* const* d_in, const int* in_sizes, int n_in,
                              void* d_out, int out_size, void* d_ws, size_t ws_size,
                              hipStream_t stream) {
  const float* x     = (const float*)d_in[0];
  const float* rel   = (const float*)d_in[1];
  const int*   elist = (const int*)d_in[2];
  const int*   etype = (const int*)d_in[3];
  const float* hW    = (const float*)d_in[4];
  const float* hWr   = (const float*)d_in[5];
  const float* ha    = (const float*)d_in[6];
  const float* hcr   = (const float*)d_in[7];
  const float* W1    = (const float*)d_in[8];
  const float* b1    = (const float*)d_in[9];
  const float* W2    = (const float*)d_in[10];
  const float* b2    = (const float*)d_in[11];
  const float* topW  = (const float*)d_in[12];
  const float* fuseW = (const float*)d_in[13];
  const float* fuseb = (const float*)d_in[14];

  const int N = in_sizes[0] / FDIM;
  const int R = in_sizes[1] / FDIM;
  const int E = in_sizes[3];
  const int* rowp = elist;
  const int* colp = elist + E;

  // workspace carve-out (256B aligned)
  char* w = (char*)d_ws; size_t off = 0;
  auto alloc = [&](size_t bytes) -> void* {
    void* p = w + off; off += (bytes + 255) & ~(size_t)255; return p;
  };
  float*    VEC    = (float*)alloc((size_t)HEADS * 3 * FDIM * 4);
  float*    c2     = (float*)alloc(HEADS * 4);
  float*    wvec   = (float*)alloc((size_t)HEADS * R * FDIM * 4);
  float*    st     = (float*)alloc((size_t)N * 8 * 4);
  float*    y      = (float*)alloc((size_t)N * HD * 4);
  float*    rowsum = (float*)alloc((size_t)N * 4 * 4);
  float*    num    = (float*)alloc((size_t)N * HD * 4);   // -> h_local
  float*    HgA    = (float*)alloc((size_t)N * HD * 4);
  float*    HgB    = (float*)alloc((size_t)N * HD * 4);
  float*    deg    = (float*)alloc((size_t)N * 4);
  float*    dinv   = (float*)alloc((size_t)N * 4);
  uint32_t* BpY    = (uint32_t*)alloc((size_t)64 * HD * 4);   // x@Wcat weights
  uint32_t* BpF    = (uint32_t*)alloc((size_t)256 * HD * 4);  // fuse weights
  uint32_t* BpT    = (uint32_t*)alloc((size_t)64 * HD * 4);   // top weights

  float* out_fused = (float*)d_out;
  float* out_rel   = (float*)d_out + (size_t)N * HD;

  // 1) precompute head vectors + per-relation wvec
  k_prevec<<<HEADS, FDIM, 0, stream>>>(hW, hWr, ha, hcr, VEC, c2);
  k_relmlp<<<dim3(R, HEADS), 256, 0, stream>>>(rel, W1, b1, W2, b2, VEC, wvec, R);

  // 2) pack bf16 weights for WMMA
  k_pack_headW<<<(64 * HD + 255) / 256, 256, 0, stream>>>(hW, BpY);
  k_pack_rowmajor<<<(256 * HD + 255) / 256, 256, 0, stream>>>(fuseW, BpF, 512, HD);
  k_pack_rowmajor<<<(64 * HD + 255) / 256, 256, 0, stream>>>(topW, BpT, FDIM, HD);

  // 3) zero accumulators (ws is poisoned by the harness)
  k_zero<<<2048, 256, 0, stream>>>(rowsum, (long)N * 4);
  k_zero<<<4096, 256, 0, stream>>>(num, (long)N * HD);
  k_zero<<<1024, 256, 0, stream>>>(deg, (long)N);

  // 4) per-node scalars + y = x @ Wcat (WMMA)
  k_node_scalars<<<(N + 7) / 8, 256, 0, stream>>>(x, VEC, st, N);
  {
    int tot = ((N + 127) / 128) * (HD / 64);
    k_wmma_gemm_t<128, 0><<<tot, 256, 0, stream>>>(x, nullptr, BpY, nullptr,
                                                   y, N, HD, 0);
  }

  // 5) degrees -> dinv
  k_count_deg<<<(E + 255) / 256, 256, 0, stream>>>(rowp, deg, E);
  k_dinv<<<(N + 255) / 256, 256, 0, stream>>>(deg, dinv, N);

  // 6) attention edges: rowsum + num accumulation (wave per edge)
  k_edges<<<(E + 7) / 8, 256, 0, stream>>>(x, rowp, colp, etype, wvec, st, c2, y,
                                           rowsum, num, E, R);
  k_finalize_local<<<(N * HD + 255) / 256, 256, 0, stream>>>(num, rowsum, N);

  // 7) K=3 propagation hops: dst = D^-1/2 (A+I) D^-1/2 src
  int nb_nc = (N * HD + 255) / 256;
  int nb_e  = (E + 7) / 8;
  k_hop_init <<<nb_nc, 256, 0, stream>>>(HgA, num, dinv, N);
  k_hop_edges<<<nb_e, 256, 0, stream>>>(rowp, colp, dinv, HgA, num, E);
  k_hop_init <<<nb_nc, 256, 0, stream>>>(HgB, HgA, dinv, N);
  k_hop_edges<<<nb_e, 256, 0, stream>>>(rowp, colp, dinv, HgB, HgA, E);
  k_hop_init <<<nb_nc, 256, 0, stream>>>(HgA, HgB, dinv, N);
  k_hop_edges<<<nb_e, 256, 0, stream>>>(rowp, colp, dinv, HgA, HgB, E);

  // 8) h_fused = elu([h_local | Hg] @ fuse_W + fuse_b)   (WMMA, split A)
  {
    int tot = ((N + 127) / 128) * (HD / 64);
    k_wmma_gemm_t<256, 256><<<tot, 256, 0, stream>>>(num, HgA, BpF, fuseb,
                                                     out_fused, N, HD, 1);
  }

  // 9) x_rel = rel @ top_W   (WMMA)
  {
    int tot = ((R + 127) / 128) * (HD / 64);
    k_wmma_gemm_t<128, 0><<<tot, 256, 0, stream>>>(rel, nullptr, BpT, nullptr,
                                                   out_rel, R, HD, 0);
  }
}
